// CorrFlowFusion_49228915147506
// MI455X (gfx1250) — matrix operand order repaired and checked by reading, not compiled, not run on hardware
//
#include <hip/hip_runtime.h>

// ---------------------------------------------------------------------------
// CDNA5 / gfx1250 implementation of CorrFlowFusion.
// Heavy N^2 (1024x1024) per-pair MLPs run as fused WMMA f16 kernels; the
// small point-set stages (FPS / ball-query / SA / FP-interp / heads) are
// plain VALU kernels (~6% of total FLOPs).
//
// Fragment builds use contiguous per-lane LDS layouts so each WMMA operand
// is materialized with 2x ds_load_b128 (instead of 16x ds_load_u16).
// ---------------------------------------------------------------------------

typedef _Float16 v16h __attribute__((ext_vector_type(16)));
typedef _Float16 v8h  __attribute__((ext_vector_type(8)));
typedef float    v8f  __attribute__((ext_vector_type(8)));

union HFrag { v16h v; _Float16 h[16]; };
union FAcc  { v8f  v; float    f[8];  };

// LDS row pitches (halves), padded +8 so b128 loads stay 16B-aligned
// (pitch*2 bytes is a multiple of 16) and banks don't alias.
#define LDK128 136   // K=128 rows
#define LDK64  72    // K=64 rows

// A-matrix fragment (16x32 f16) from row-major W[M][ldk] in LDS.
// Per ISA 7.12.2, lane holds row M=lane%16; half h -> K:
//   lanes 0-15 : {kbase+0..7,  kbase+16..23}
//   lanes 16-31: {kbase+8..15, kbase+24..31}
// -> two contiguous 8-half runs -> 2x ds_load_b128.
__device__ __forceinline__ v16h wmma_a_frag(const _Float16* __restrict__ W,
                                            int m, int kbase, int ldk, int lane) {
  const int hb = (lane >= 16) ? 8 : 0;
  HFrag a;
  *(v8h*)&a.h[0] = *(const v8h*)(W + m * ldk + kbase + hb);
  *(v8h*)&a.h[8] = *(const v8h*)(W + m * ldk + kbase + 16 + hb);
  return a.v;
}

// B-matrix fragment (32x16 f16) from pair-major activation H[col][ldk] in LDS.
// lane: col = lane%16, K same as A: {kbase+hb+0..7, kbase+16+hb+0..7}, hb = lane>=16 ? 8 : 0
// -> two contiguous 8-half runs -> 2x ds_load_b128.
__device__ __forceinline__ v16h wmma_b_frag(const _Float16* __restrict__ H,
                                            int kbase, int col, int ldk, int lane) {
  const int off = (lane >= 16) ? 8 : 0;
  HFrag b;
  *(v8h*)&b.h[0] = *(const v8h*)(H + col * ldk + kbase + off);
  *(v8h*)&b.h[8] = *(const v8h*)(H + col * ldk + kbase + off + 16);
  return b.v;
}

// Order-preserving float->uint key for atomic max of signed floats.
__device__ __forceinline__ unsigned ordkey(float x) {
  unsigned u = __float_as_uint(x);
  return (u & 0x80000000u) ? ~u : (u | 0x80000000u);
}
__device__ __forceinline__ float inv_ordkey(unsigned k) {
  return __uint_as_float((k & 0x80000000u) ? (k & 0x7fffffffu) : ~k);
}

// ---------------------------------------------------------------------------
// mask1 layer-1 collapse: A[c][i] = W1[c, :64] @ feat1[:, i] + b1[c]
//                         Bc[c][i] = W1[c, 64:] @ feat2[:, i]
// ---------------------------------------------------------------------------
__global__ void k_mask1_pre(const float* __restrict__ W1, const float* __restrict__ b1,
                            const float* __restrict__ f1, const float* __restrict__ f2,
                            float* __restrict__ A, float* __restrict__ Bc) {
  int g = blockIdx.x * blockDim.x + threadIdx.x;
  if (g >= 128 * 1024) return;
  int c = g >> 10, i = g & 1023;
  float sa = b1[c], sb = 0.f;
  for (int f = 0; f < 64; ++f) {
    sa += W1[c * 128 + f]      * f1[f * 1024 + i];
    sb += W1[c * 128 + 64 + f] * f2[f * 1024 + i];
  }
  A[g] = sa; Bc[g] = sb;
}

// ---------------------------------------------------------------------------
// Fused mask pipeline over all (i,j):
//   h1 = relu(A[:,i] + Bc[:,j])                 (128)
//   h2 = relu(W2 @ h1 + b2)                     (64)   <- WMMA f16
//   coarse[i,j] = Wm . h2 + bm
//   fpm[c,j]    = max_i h2[c]
// Workgroup: 8 waves, owns 2 j columns, loops i in chunks of 16.
// ---------------------------------------------------------------------------
__global__ __launch_bounds__(256) void k_mask_fused(
    const float* __restrict__ A128, const float* __restrict__ B128,
    const float* __restrict__ W2,  const float* __restrict__ b2,
    const float* __restrict__ Wm,  const float* __restrict__ bm,
    float* __restrict__ coarse, float* __restrict__ fpm) {
  __shared__ __align__(16) _Float16 sW2[64 * LDK128];
  __shared__ __align__(16) _Float16 sH1[2][16 * LDK128];  // [j][pair][K]
  __shared__ float    sB[2][128];
  __shared__ float    sb2[64];
  __shared__ float    sWm[64];
  __shared__ float    sCoarse[2][16];
  __shared__ unsigned sFpm[2][64];

  const int tid  = threadIdx.x;
  const int lane = tid & 31;
  const int wave = tid >> 5;
  const int jl   = wave >> 2;   // which of the 2 j columns
  const int mt   = wave & 3;    // m-tile (16 rows of W2 each)
  const int j0   = blockIdx.x * 2;
  const float bm0 = bm[0];

  for (int v = tid; v < 64 * 128; v += 256) {
    int m = v >> 7, c = v & 127;
    sW2[m * LDK128 + c] = (_Float16)W2[v];
  }
  for (int v = tid; v < 2 * 128; v += 256) {
    int jj = v >> 7, c = v & 127;
    sB[jj][c] = B128[c * 1024 + j0 + jj];
  }
  for (int v = tid; v < 64; v += 256) { sb2[v] = b2[v]; sWm[v] = Wm[v]; }
  for (int v = tid; v < 128; v += 256) sFpm[v >> 6][v & 63] = 0u;  // h2 >= 0
  __syncthreads();

  // Loop-invariant A-operand fragments: W2 rows [mt*16 .. mt*16+15], K=128.
  v16h aW[4];
#pragma unroll
  for (int k = 0; k < 4; ++k)
    aW[k] = wmma_a_frag(sW2, mt * 16 + (lane & 15), k * 32, LDK128, lane);

  for (int ic = 0; ic < 64; ++ic) {
    const int i0 = ic << 4;
    __syncthreads();
    if (tid < 32) sCoarse[tid >> 4][tid & 15] = 0.f;
    // h1 = relu(A[:,i] + B[:,j]) in f16 -> LDS (pair-major)
    for (int v = tid; v < 2 * 128 * 16; v += 256) {
      int jj = v >> 11, rem = v & 2047, c = rem >> 4, ii = rem & 15;
      float x = A128[c * 1024 + i0 + ii] + sB[jj][c];
      sH1[jj][ii * LDK128 + c] = (_Float16)fmaxf(x, 0.f);
    }
    if (ic < 63 && tid < 128)
      __builtin_prefetch(A128 + tid * 1024 + i0 + 16, 0, 1);  // global_prefetch
    __syncthreads();

    FAcc acc; acc.v = (v8f){0.f, 0.f, 0.f, 0.f, 0.f, 0.f, 0.f, 0.f};
#pragma unroll
    for (int k = 0; k < 4; ++k) {
      v16h bf = wmma_b_frag(sH1[jl], k * 32, lane & 15, LDK128, lane);
      acc.v = __builtin_amdgcn_wmma_f32_16x16x32_f16(
          false, aW[k], false, bf, (short)0, acc.v, false, false);
    }
    // D layout: VGPR r -> M = r (+8 for lanes 16-31); N = lane%16.
    float part = 0.f;
    const int mbase = mt * 16 + ((lane >= 16) ? 8 : 0);
#pragma unroll
    for (int r = 0; r < 8; ++r) {
      const int m = mbase + r;
      float h2 = fmaxf(acc.f[r] + sb2[m], 0.f);
      part += sWm[m] * h2;
      atomicMax(&sFpm[jl][m], __float_as_uint(h2));  // h2 >= 0: bits monotone
    }
    atomicAdd(&sCoarse[jl][lane & 15], part);
    __syncthreads();
    if (tid < 32) {
      int jj = tid >> 4, ii = tid & 15;
      coarse[(size_t)(i0 + ii) * 1024 + (j0 + jj)] = sCoarse[jj][ii] + bm0;
    }
  }
  __syncthreads();
  for (int v = tid; v < 128; v += 256)
    fpm[(size_t)(v & 63) * 1024 + j0 + (v >> 6)] = __uint_as_float(sFpm[v >> 6][v & 63]);
}

// ---------------------------------------------------------------------------
// mask2 + conv_prob: prob[j] from fpm[:, j]
// ---------------------------------------------------------------------------
__global__ void k_prob(const float* __restrict__ fpm,
                       const float* __restrict__ W2a, const float* __restrict__ b2a,
                       const float* __restrict__ W2b, const float* __restrict__ b2b,
                       const float* __restrict__ Wp,  const float* __restrict__ bp,
                       float* __restrict__ prob) {
  int j = blockIdx.x * blockDim.x + threadIdx.x;
  if (j >= 1024) return;
  float t1[64], t2[64];
  for (int o = 0; o < 64; ++o) {
    float s = b2a[o];
    for (int c = 0; c < 64; ++c) s += W2a[o * 64 + c] * fpm[c * 1024 + j];
    t1[o] = fmaxf(s, 0.f);
  }
  for (int o = 0; o < 64; ++o) {
    float s = b2b[o];
    for (int c = 0; c < 64; ++c) s += W2b[o * 64 + c] * t1[c];
    t2[o] = s;
  }
  float p = bp[0];
  for (int c = 0; c < 64; ++c) p += Wp[c] * t2[c];
  prob[j] = p;
}

// refined[i,j] = coarse[i,j] * prob[i]  -> first output segment
__global__ void k_refined(const float* __restrict__ coarse, const float* __restrict__ prob,
                          float* __restrict__ out) {
  size_t g = (size_t)blockIdx.x * blockDim.x + threadIdx.x;
  if (g >= (size_t)1024 * 1024) return;
  out[g] = coarse[g] * prob[g >> 10];
}

// ---------------------------------------------------------------------------
// Fused pnd pipeline over all (i,j):
//   x0 = [pc2[j]-pc1[i] (3), coarse[i,j]*prob[i]]
//   h1 = relu(W1 @ x0 + b1)    (64, VALU; K=4)
//   h2 = relu(W2 @ h1 + b2)    (64, WMMA)
//   h3 = W3 @ h2 + b3          (128, WMMA)
//   feat_corr[c,j] = max_i h3[c]
// Workgroup: 4 waves, owns 1 j, loops i in chunks of 16.
// ---------------------------------------------------------------------------
__global__ __launch_bounds__(128) void k_pnd_fused(
    const float* __restrict__ pc1, const float* __restrict__ pc2,
    const float* __restrict__ coarse, const float* __restrict__ prob,
    const float* __restrict__ W1, const float* __restrict__ b1,
    const float* __restrict__ W2, const float* __restrict__ b2,
    const float* __restrict__ W3, const float* __restrict__ b3,
    float* __restrict__ feat_corr) {
  __shared__ __align__(16) _Float16 sW2[64 * LDK64];
  __shared__ __align__(16) _Float16 sW3[128 * LDK64];
  __shared__ __align__(16) _Float16 sH1[16 * LDK64];   // [pair][K]
  __shared__ __align__(16) _Float16 sH2[16 * LDK64];   // [pair][K]
  __shared__ float    sW1[64 * 4], sb1[64], sb2[64], sb3[128];
  __shared__ float    sX0[16][4];
  __shared__ unsigned sFC[128];
  __shared__ float    sP2[3];

  const int tid  = threadIdx.x;
  const int lane = tid & 31;
  const int w    = tid >> 5;   // wave 0..3
  const int j    = blockIdx.x;

  for (int v = tid; v < 64 * 64;  v += 128) {
    int m = v >> 6, c = v & 63;
    sW2[m * LDK64 + c] = (_Float16)W2[v];
  }
  for (int v = tid; v < 128 * 64; v += 128) {
    int m = v >> 6, c = v & 63;
    sW3[m * LDK64 + c] = (_Float16)W3[v];
  }
  for (int v = tid; v < 64 * 4;   v += 128) sW1[v] = W1[v];
  for (int v = tid; v < 64;       v += 128) { sb1[v] = b1[v]; sb2[v] = b2[v]; }
  for (int v = tid; v < 128;      v += 128) { sb3[v] = b3[v]; sFC[v] = ordkey(-3e38f); }
  if (tid < 3) sP2[tid] = pc2[j * 3 + tid];
  __syncthreads();

  v16h aW2[2], aW3a[2], aW3b[2];
#pragma unroll
  for (int k = 0; k < 2; ++k) {
    aW2[k]  = wmma_a_frag(sW2, w * 16       + (lane & 15), k * 32, LDK64, lane);
    aW3a[k] = wmma_a_frag(sW3, w * 16       + (lane & 15), k * 32, LDK64, lane);
    aW3b[k] = wmma_a_frag(sW3, (w + 4) * 16 + (lane & 15), k * 32, LDK64, lane);
  }

  for (int ic = 0; ic < 64; ++ic) {
    const int i0 = ic << 4;
    __syncthreads();
    if (tid < 16) {
      int i = i0 + tid;
      sX0[tid][0] = sP2[0] - pc1[i * 3 + 0];
      sX0[tid][1] = sP2[1] - pc1[i * 3 + 1];
      sX0[tid][2] = sP2[2] - pc1[i * 3 + 2];
      sX0[tid][3] = coarse[(size_t)i * 1024 + j] * prob[i];
    }
    __syncthreads();
    for (int v = tid; v < 64 * 16; v += 128) {
      int c = v >> 4, ii = v & 15;
      float s = sb1[c];
#pragma unroll
      for (int q = 0; q < 4; ++q) s += sW1[c * 4 + q] * sX0[ii][q];
      sH1[ii * LDK64 + c] = (_Float16)fmaxf(s, 0.f);
    }
    __syncthreads();

    // layer 2 (WMMA, K=64)
    FAcc acc; acc.v = (v8f){0.f, 0.f, 0.f, 0.f, 0.f, 0.f, 0.f, 0.f};
#pragma unroll
    for (int k = 0; k < 2; ++k) {
      v16h bf = wmma_b_frag(sH1, k * 32, lane & 15, LDK64, lane);
      acc.v = __builtin_amdgcn_wmma_f32_16x16x32_f16(
          false, aW2[k], false, bf, (short)0, acc.v, false, false);
    }
    const int mloc = ((lane >= 16) ? 8 : 0);
#pragma unroll
    for (int r = 0; r < 8; ++r) {
      int m = w * 16 + mloc + r;
      sH2[(lane & 15) * LDK64 + m] = (_Float16)fmaxf(acc.f[r] + sb2[m], 0.f);
    }
    __syncthreads();

    // layer 3 (WMMA, M=128 -> two m-tiles per wave; shared B fragments)
    v16h bf3[2];
#pragma unroll
    for (int k = 0; k < 2; ++k)
      bf3[k] = wmma_b_frag(sH2, k * 32, lane & 15, LDK64, lane);

    FAcc a3; a3.v = (v8f){0.f, 0.f, 0.f, 0.f, 0.f, 0.f, 0.f, 0.f};
#pragma unroll
    for (int k = 0; k < 2; ++k)
      a3.v = __builtin_amdgcn_wmma_f32_16x16x32_f16(
          false, aW3a[k], false, bf3[k], (short)0, a3.v, false, false);
#pragma unroll
    for (int r = 0; r < 8; ++r) {
      int m = w * 16 + mloc + r;
      atomicMax(&sFC[m], ordkey(a3.f[r] + sb3[m]));
    }
    FAcc a4; a4.v = (v8f){0.f, 0.f, 0.f, 0.f, 0.f, 0.f, 0.f, 0.f};
#pragma unroll
    for (int k = 0; k < 2; ++k)
      a4.v = __builtin_amdgcn_wmma_f32_16x16x32_f16(
          false, aW3b[k], false, bf3[k], (short)0, a4.v, false, false);
#pragma unroll
    for (int r = 0; r < 8; ++r) {
      int m = (w + 4) * 16 + mloc + r;
      atomicMax(&sFC[m], ordkey(a4.f[r] + sb3[m]));
    }
  }
  __syncthreads();
  for (int v = tid; v < 128; v += 128)
    feat_corr[(size_t)v * 1024 + j] = inv_ordkey(sFC[v]);
}

// ---------------------------------------------------------------------------
// Farthest point sampling (single workgroup, sequential iterations)
// ---------------------------------------------------------------------------
__global__ __launch_bounds__(256) void k_fps(const float* __restrict__ xyz, int npts,
                                             int npoint, int* __restrict__ fidx,
                                             float* __restrict__ new_xyz) {
  __shared__ float dist[1024];
  __shared__ float rv[256];
  __shared__ int   ri[256];
  __shared__ int   sfar;
  const int tid = threadIdx.x;
  for (int v = tid; v < npts; v += 256) dist[v] = 1e10f;
  if (tid == 0) sfar = 0;
  __syncthreads();
  for (int it = 0; it < npoint; ++it) {
    const int far = sfar;
    if (tid == 0) {
      fidx[it] = far;
      new_xyz[it * 3 + 0] = xyz[far * 3 + 0];
      new_xyz[it * 3 + 1] = xyz[far * 3 + 1];
      new_xyz[it * 3 + 2] = xyz[far * 3 + 2];
    }
    const float cx = xyz[far * 3], cy = xyz[far * 3 + 1], cz = xyz[far * 3 + 2];
    float bv = -1.f; int bi = 0;
    for (int v = tid; v < npts; v += 256) {
      float dx = xyz[v * 3] - cx, dy = xyz[v * 3 + 1] - cy, dz = xyz[v * 3 + 2] - cz;
      float nd = fminf(dist[v], dx * dx + dy * dy + dz * dz);
      dist[v] = nd;
      if (nd > bv) { bv = nd; bi = v; }
    }
    rv[tid] = bv; ri[tid] = bi;
    __syncthreads();
    for (int s = 128; s > 0; s >>= 1) {
      if (tid < s) {
        if (rv[tid + s] > rv[tid] || (rv[tid + s] == rv[tid] && ri[tid + s] < ri[tid])) {
          rv[tid] = rv[tid + s]; ri[tid] = ri[tid + s];
        }
      }
      __syncthreads();
    }
    if (tid == 0) sfar = ri[0];
    __syncthreads();
  }
}

// Ball query: first nsample indices (ascending) with d<=r2, padded with first hit.
__global__ void k_ballq(const float* __restrict__ xyz, int npts,
                        const float* __restrict__ new_xyz, int npoint,
                        float r2, int nsample, int* __restrict__ nidx) {
  int s = blockIdx.x * blockDim.x + threadIdx.x;
  if (s >= npoint) return;
  float cx = new_xyz[s * 3], cy = new_xyz[s * 3 + 1], cz = new_xyz[s * 3 + 2];
  int cnt = 0, first = -1;
  for (int p = 0; p < npts && cnt < nsample; ++p) {
    float dx = xyz[p * 3] - cx, dy = xyz[p * 3 + 1] - cy, dz = xyz[p * 3 + 2] - cz;
    if (dx * dx + dy * dy + dz * dz <= r2) {
      if (first < 0) first = p;
      nidx[s * nsample + cnt++] = p;
    }
  }
  if (first < 0) first = 0;
  for (; cnt < nsample; ++cnt) nidx[s * nsample + cnt] = first;
}

// SA: 3-layer grouped MLP + max over samples. One workgroup per centroid.
__global__ __launch_bounds__(256) void k_sa_mlp(
    const float* __restrict__ xyz, int npts,
    const float* __restrict__ new_xyz, const int* __restrict__ nidx, int nsample,
    const float* __restrict__ feats, int cin,
    const float* __restrict__ W0, const float* __restrict__ b0, int d1,
    const float* __restrict__ W1, const float* __restrict__ b1, int d2,
    const float* __restrict__ W2, const float* __restrict__ b2, int d3,
    float* __restrict__ out, int npoint) {
  __shared__ float vin[272];
  __shared__ float v1[256];
  __shared__ float v2[256];
  __shared__ float vmax[512];
  const int s = blockIdx.x, tid = threadIdx.x;
  const int cin0 = cin + 3;
  for (int v = tid; v < d3; v += 256) vmax[v] = -1e30f;
  for (int k = 0; k < nsample; ++k) {
    __syncthreads();
    const int idx = nidx[s * nsample + k];
    if (tid < 3) vin[tid] = xyz[idx * 3 + tid] - new_xyz[s * 3 + tid];
    for (int v = tid; v < cin; v += 256) vin[3 + v] = feats[v * npts + idx];
    __syncthreads();
    for (int o = tid; o < d1; o += 256) {
      float acc = b0[o];
      for (int c = 0; c < cin0; ++c) acc += W0[o * cin0 + c] * vin[c];
      v1[o] = fmaxf(acc, 0.f);
    }
    __syncthreads();
    for (int o = tid; o < d2; o += 256) {
      float acc = b1[o];
      for (int c = 0; c < d1; ++c) acc += W1[o * d1 + c] * v1[c];
      v2[o] = fmaxf(acc, 0.f);
    }
    __syncthreads();
    for (int o = tid; o < d3; o += 256) {
      float acc = b2[o];
      for (int c = 0; c < d2; ++c) acc += W2[o * d2 + c] * v2[c];
      vmax[o] = fmaxf(vmax[o], fmaxf(acc, 0.f));
    }
  }
  __syncthreads();
  for (int o = tid; o < d3; o += 256) out[o * npoint + s] = vmax[o];
}

// FP: 3-NN inverse-distance interpolation + optional unknown-feature concat.
__global__ void k_fp_interp(const float* __restrict__ unk_xyz, int n_unk,
                            const float* __restrict__ kn_xyz, int n_kn,
                            const float* __restrict__ kn_feats, int ck,
                            const float* __restrict__ unk_feats, int cu,
                            float* __restrict__ hout) {
  int p = blockIdx.x * blockDim.x + threadIdx.x;
  if (p >= n_unk) return;
  float px = unk_xyz[p * 3], py = unk_xyz[p * 3 + 1], pz = unk_xyz[p * 3 + 2];
  float d0 = 1e30f, d1 = 1e30f, d2 = 1e30f;
  int i0 = 0, i1 = 0, i2 = 0;
  for (int q = 0; q < n_kn; ++q) {
    float dx = px - kn_xyz[q * 3], dy = py - kn_xyz[q * 3 + 1], dz = pz - kn_xyz[q * 3 + 2];
    float d = dx * dx + dy * dy + dz * dz;
    if (d < d0)      { d2 = d1; i2 = i1; d1 = d0; i1 = i0; d0 = d; i0 = q; }
    else if (d < d1) { d2 = d1; i2 = i1; d1 = d; i1 = q; }
    else if (d < d2) { d2 = d; i2 = q; }
  }
  float w0 = 1.f / (d0 + 1e-8f), w1 = 1.f / (d1 + 1e-8f), w2 = 1.f / (d2 + 1e-8f);
  float ws = w0 + w1 + w2;
  w0 /= ws; w1 /= ws; w2 /= ws;
  for (int c = 0; c < cu; ++c) hout[c * n_unk + p] = unk_feats[c * n_unk + p];
  for (int c = 0; c < ck; ++c)
    hout[(cu + c) * n_unk + p] = w0 * kn_feats[c * n_kn + i0] +
                                 w1 * kn_feats[c * n_kn + i1] +
                                 w2 * kn_feats[c * n_kn + i2];
}

// Generic conv1x1 (out[o][p] = act(W[o,:] . in[:,p] + b[o]))
__global__ void k_conv1x1(const float* __restrict__ in, int cin, int np,
                          const float* __restrict__ W, const float* __restrict__ b,
                          int cout, float* __restrict__ out, int dorelu) {
  int g = blockIdx.x * blockDim.x + threadIdx.x;
  if (g >= cout * np) return;
  int o = g / np, p = g - o * np;
  float acc = b[o];
  const float* w = W + o * cin;
  for (int c = 0; c < cin; ++c) acc += w[c] * in[c * np + p];
  out[g] = dorelu ? fmaxf(acc, 0.f) : acc;
}

// x[c][p] = pc1[p][c] + flow[c][p]
__global__ void k_addT(const float* __restrict__ pc1, const float* __restrict__ flow,
                       float* __restrict__ x) {
  int g = blockIdx.x * blockDim.x + threadIdx.x;
  if (g >= 3 * 1024) return;
  int c = g >> 10, p = g & 1023;
  x[g] = pc1[p * 3 + c] + flow[g];
}

// transpose (256,1024) -> (1024,256)
__global__ void k_transpose_out(const float* __restrict__ xin, float* __restrict__ out) {
  int g = blockIdx.x * blockDim.x + threadIdx.x;
  if (g >= 256 * 1024) return;
  int c = g >> 10, p = g & 1023;
  out[p * 256 + c] = xin[g];
}

// ---------------------------------------------------------------------------
// Host side
// ---------------------------------------------------------------------------
// Assumed d_in flatten order: pc_1, pc_2, pc_feat_1, pc_feat_2, then params
// as a pytree (keys sorted), each layer (W, b).
enum {
  IN_PC1 = 0, IN_PC2, IN_F1, IN_F2,
  P_CM_W, P_CM_B, P_CP_W, P_CP_B,
  P_FLOW0_W, P_FLOW0_B, P_FLOW1_W, P_FLOW1_B, P_FLOW2_W, P_FLOW2_B,
  P_FP0_0_W, P_FP0_0_B, P_FP0_1_W, P_FP0_1_B,
  P_FP1_0_W, P_FP1_0_B, P_FP1_1_W, P_FP1_1_B,
  P_FP2_0_W, P_FP2_0_B, P_FP2_1_W, P_FP2_1_B,
  P_M1_0_W, P_M1_0_B, P_M1_1_W, P_M1_1_B,
  P_M2_0_W, P_M2_0_B, P_M2_1_W, P_M2_1_B,
  P_PND0_W, P_PND0_B, P_PND1_W, P_PND1_B, P_PND2_W, P_PND2_B,
  P_PNF0_W, P_PNF0_B, P_PNF1_W, P_PNF1_B, P_PNF2_W, P_PNF2_B,
  P_SA1_0_W, P_SA1_0_B, P_SA1_1_W, P_SA1_1_B, P_SA1_2_W, P_SA1_2_B,
  P_SA2_0_W, P_SA2_0_B, P_SA2_1_W, P_SA2_1_B, P_SA2_2_W, P_SA2_2_B
};

extern "C" void kernel_launch(void* const* d_in, const int* in_sizes, int n_in,
                              void* d_out, int out_size, void* d_ws, size_t ws_size,
                              hipStream_t stream) {
  (void)in_sizes; (void)n_in; (void)out_size; (void)ws_size;
  auto F = [&](int i) { return (const float*)d_in[i]; };

  float* wsf = (float*)d_ws;
  size_t off = 0;
  auto alloc = [&](size_t n) { float* p = wsf + off; off += (n + 63) & ~(size_t)63; return p; };

  float* A128      = alloc(128 * 1024);
  float* B128      = alloc(128 * 1024);
  float* coarse    = alloc((size_t)1024 * 1024);
  float* fpm       = alloc(64 * 1024);
  float* prob      = alloc(1024);
  float* feat_corr = alloc(128 * 1024);
  int*   fidx1     = (int*)alloc(256);
  float* new_xyz1  = alloc(256 * 3);
  int*   nidx1     = (int*)alloc(256 * 32);
  float* l2_points = alloc(256 * 256);
  int*   fidx2     = (int*)alloc(64);
  float* l3_xyz    = alloc(64 * 3);
  int*   nidx2     = (int*)alloc(64 * 32);
  float* l3_points = alloc(512 * 64);
  float* fp0_h     = alloc(768 * 256);
  float* fp0_t1    = alloc(256 * 256);
  float* fp0_out   = alloc(256 * 256);
  float* fp1_h     = alloc(384 * 1024);
  float* fp1_t1    = alloc(256 * 1024);
  float* fp1_out   = alloc(128 * 1024);
  float* fp2_h     = alloc(128 * 1024);
  float* fp2_t1    = alloc(128 * 1024);
  float* l0_points = alloc(128 * 1024);
  float* g1        = alloc(128 * 1024);
  float* g2        = alloc(64 * 1024);
  float* x_in      = alloc(3 * 1024);
  float* pnf1      = alloc(64 * 1024);
  float* pnf2      = alloc(128 * 1024);
  float* pnf3      = alloc(256 * 1024);

  float* out_refined = (float*)d_out;                       // 1024*1024
  float* out_flow    = out_refined + (size_t)1024 * 1024;   // 3*1024
  float* out_x       = out_flow + 3 * 1024;                 // 1024*256

  const float* pc1 = F(IN_PC1);
  const float* pc2 = F(IN_PC2);

  // --- mask pipeline (layer-1 collapse + fused WMMA N^2 kernel) ---
  k_mask1_pre<<<512, 256, 0, stream>>>(F(P_M1_0_W), F(P_M1_0_B), F(IN_F1), F(IN_F2),
                                       A128, B128);
  k_mask_fused<<<512, 256, 0, stream>>>(A128, B128, F(P_M1_1_W), F(P_M1_1_B),
                                        F(P_CM_W), F(P_CM_B), coarse, fpm);
  k_prob<<<4, 256, 0, stream>>>(fpm, F(P_M2_0_W), F(P_M2_0_B), F(P_M2_1_W), F(P_M2_1_B),
                                F(P_CP_W), F(P_CP_B), prob);
  k_refined<<<4096, 256, 0, stream>>>(coarse, prob, out_refined);

  // --- pnd pipeline (fused WMMA N^2 kernel with max over i) ---
  k_pnd_fused<<<1024, 128, 0, stream>>>(pc1, pc2, coarse, prob,
                                        F(P_PND0_W), F(P_PND0_B),
                                        F(P_PND1_W), F(P_PND1_B),
                                        F(P_PND2_W), F(P_PND2_B), feat_corr);

  // --- SA1 ---
  k_fps<<<1, 256, 0, stream>>>(pc1, 1024, 256, fidx1, new_xyz1);
  k_ballq<<<1, 256, 0, stream>>>(pc1, 1024, new_xyz1, 256, 0.04f, 32, nidx1);
  k_sa_mlp<<<256, 256, 0, stream>>>(pc1, 1024, new_xyz1, nidx1, 32, feat_corr, 128,
                                    F(P_SA1_0_W), F(P_SA1_0_B), 128,
                                    F(P_SA1_1_W), F(P_SA1_1_B), 128,
                                    F(P_SA1_2_W), F(P_SA1_2_B), 256,
                                    l2_points, 256);
  // --- SA2 ---
  k_fps<<<1, 256, 0, stream>>>(new_xyz1, 256, 64, fidx2, l3_xyz);
  k_ballq<<<1, 256, 0, stream>>>(new_xyz1, 256, l3_xyz, 64, 0.16f, 32, nidx2);
  k_sa_mlp<<<64, 256, 0, stream>>>(new_xyz1, 256, l3_xyz, nidx2, 32, l2_points, 256,
                                   F(P_SA2_0_W), F(P_SA2_0_B), 256,
                                   F(P_SA2_1_W), F(P_SA2_1_B), 256,
                                   F(P_SA2_2_W), F(P_SA2_2_B), 512,
                                   l3_points, 64);

  // --- FP0: l2 <- (l2, l3) ---
  k_fp_interp<<<1, 256, 0, stream>>>(new_xyz1, 256, l3_xyz, 64, l3_points, 512,
                                     l2_points, 256, fp0_h);
  k_conv1x1<<<(256 * 256 + 255) / 256, 256, 0, stream>>>(fp0_h, 768, 256,
      F(P_FP0_0_W), F(P_FP0_0_B), 256, fp0_t1, 1);
  k_conv1x1<<<(256 * 256 + 255) / 256, 256, 0, stream>>>(fp0_t1, 256, 256,
      F(P_FP0_1_W), F(P_FP0_1_B), 256, fp0_out, 1);

  // --- FP1: l1 <- (pc1/feat_corr, l2) ---
  k_fp_interp<<<4, 256, 0, stream>>>(pc1, 1024, new_xyz1, 256, fp0_out, 256,
                                     feat_corr, 128, fp1_h);
  k_conv1x1<<<(256 * 1024 + 255) / 256, 256, 0, stream>>>(fp1_h, 384, 1024,
      F(P_FP1_0_W), F(P_FP1_0_B), 256, fp1_t1, 1);
  k_conv1x1<<<(128 * 1024 + 255) / 256, 256, 0, stream>>>(fp1_t1, 256, 1024,
      F(P_FP1_1_W), F(P_FP1_1_B), 128, fp1_out, 1);

  // --- FP2: l0 <- (pc1, l1) with no unknown feats ---
  k_fp_interp<<<4, 256, 0, stream>>>(pc1, 1024, pc1, 1024, fp1_out, 128,
                                     nullptr, 0, fp2_h);
  k_conv1x1<<<(128 * 1024 + 255) / 256, 256, 0, stream>>>(fp2_h, 128, 1024,
      F(P_FP2_0_W), F(P_FP2_0_B), 128, fp2_t1, 1);
  k_conv1x1<<<(128 * 1024 + 255) / 256, 256, 0, stream>>>(fp2_t1, 128, 1024,
      F(P_FP2_1_W), F(P_FP2_1_B), 128, l0_points, 1);

  // --- flow head ---
  k_conv1x1<<<(128 * 1024 + 255) / 256, 256, 0, stream>>>(l0_points, 128, 1024,
      F(P_FLOW0_W), F(P_FLOW0_B), 128, g1, 1);
  k_conv1x1<<<(64 * 1024 + 255) / 256, 256, 0, stream>>>(g1, 128, 1024,
      F(P_FLOW1_W), F(P_FLOW1_B), 64, g2, 1);
  k_conv1x1<<<(3 * 1024 + 255) / 256, 256, 0, stream>>>(g2, 64, 1024,
      F(P_FLOW2_W), F(P_FLOW2_B), 3, out_flow, 0);

  // --- pnf head on warped points ---
  k_addT<<<12, 256, 0, stream>>>(pc1, out_flow, x_in);
  k_conv1x1<<<(64 * 1024 + 255) / 256, 256, 0, stream>>>(x_in, 3, 1024,
      F(P_PNF0_W), F(P_PNF0_B), 64, pnf1, 1);
  k_conv1x1<<<(128 * 1024 + 255) / 256, 256, 0, stream>>>(pnf1, 64, 1024,
      F(P_PNF1_W), F(P_PNF1_B), 128, pnf2, 1);
  k_conv1x1<<<(256 * 1024 + 255) / 256, 256, 0, stream>>>(pnf2, 128, 1024,
      F(P_PNF2_W), F(P_PNF2_B), 256, pnf3, 0);
  k_transpose_out<<<(256 * 1024 + 255) / 256, 256, 0, stream>>>(pnf3, out_x);
}
